// CLIPGNN_76751065579704
// MI455X (gfx1250) — compile-verified, matrix operand order checked by reading, not run on hardware
//
#include <hip/hip_runtime.h>

// ---------------------------------------------------------------------------
// CDNA5 (gfx1250) GCN forward:  x -> GCNConv(W1) -> ReLU -> GCNConv(W2) -> ReLU -> Linear(Wc)
// GEMMs run on v_wmma_f32_16x16x32_bf16 (bf16 in, f32 accum), 32x32 C block
// per wave (4 accumulators) for 2x operand reuse per WMMA.
// Aggregation uses float4 gathers + global_atomic_add_f32 scatter.
// ---------------------------------------------------------------------------

typedef __attribute__((ext_vector_type(16))) __bf16 v16bf;
typedef __attribute__((ext_vector_type(8)))  float  v8f;

struct U4 { unsigned int x, y, z, w; };          // POD 128-bit chunk (union-safe)
union FragAB { v16bf v; U4 q[2]; };              // 16 bf16 = 32 bytes = 2 x b128 loads

#define FEAT 512                                  // CLIP_DIM == HIDDEN == 512

__device__ __forceinline__ unsigned short f2bf(float f) {
    unsigned int u = __float_as_uint(f);
    u += 0x7FFFu + ((u >> 16) & 1u);              // round-to-nearest-even
    return (unsigned short)(u >> 16);
}

// ---------------- degree / normalization ----------------
__global__ void deg_init_k(float* deg, int n) {
    int i = blockIdx.x * blockDim.x + threadIdx.x;
    if (i < n) deg[i] = 1.0f;                     // self-loop
}
__global__ void deg_edges_k(const int* __restrict__ dst, float* deg, int e) {
    int i = blockIdx.x * blockDim.x + threadIdx.x;
    if (i < e) atomicAdd(&deg[dst[i]], 1.0f);
}
__global__ void norm_k(const float* __restrict__ deg, float* nrm, int n) {
    int i = blockIdx.x * blockDim.x + threadIdx.x;
    if (i < n) nrm[i] = rsqrtf(deg[i]);
}

// ---------------- f32 -> bf16 conversion ----------------
__global__ void f32_to_bf16_k(const float* __restrict__ in, unsigned short* __restrict__ out,
                              size_t n) {
    size_t i = (size_t)blockIdx.x * blockDim.x + threadIdx.x;
    if (i < n) out[i] = f2bf(in[i]);
}
// weight convert with zero-padded rows (for the classifier N=100 -> 128)
__global__ void conv_pad_k(const float* __restrict__ in, unsigned short* __restrict__ out,
                           int rows, int cols, int padRows) {
    size_t i = (size_t)blockIdx.x * blockDim.x + threadIdx.x;
    size_t total = (size_t)padRows * cols;
    if (i >= total) return;
    int r = (int)(i / cols);
    out[i] = (r < rows) ? f2bf(in[i]) : (unsigned short)0;
}

// ---------------- bf16 WMMA GEMM: C[M,Nout] = A[M,K] * B[Npad,K]^T (+bias) ----------------
// Each wave computes a 32x32 C block: 2 A-frags + 2 B-frags -> 4 WMMAs per k-step.
// Block = 256 threads = 8 waves arranged 2(M) x 4(N): block covers 64 rows x 128 cols.
// grid.x = ceil(M/64), grid.y = Npad/128.
__global__ void gemm_bf16_wmma_k(const unsigned short* __restrict__ A,
                                 const unsigned short* __restrict__ B,
                                 float* __restrict__ C,
                                 const float* __restrict__ bias,
                                 int M, int K, int ldc, int Nout) {
    const int lane  = threadIdx.x & 31;
    const int wave  = threadIdx.x >> 5;
    const int r     = lane & 15;
    const int hl    = lane >> 4;                   // lane half: selects K sub-block
    const int waveN = wave & 3;
    const int waveM = wave >> 2;
    const int mBase = blockIdx.x * 64 + waveM * 32;
    const int nBase = (blockIdx.y * 4 + waveN) * 32;

    // A rows clamped to M-1 (edge tile): duplicated rows only affect their own
    // (store-masked) C rows. B rows always in range (Npad multiple of 32).
    int rA0 = mBase + r;      if (rA0 > M - 1) rA0 = M - 1;
    int rA1 = mBase + 16 + r; if (rA1 > M - 1) rA1 = M - 1;
    const unsigned short* aRow0 = A + (size_t)rA0 * K;
    const unsigned short* aRow1 = A + (size_t)rA1 * K;
    const unsigned short* bRow0 = B + (size_t)(nBase + r) * K;
    const unsigned short* bRow1 = B + (size_t)(nBase + 16 + r) * K;

    v8f acc00 = {}, acc01 = {}, acc10 = {}, acc11 = {};
    for (int k0 = 0; k0 < K; k0 += 32) {
        __builtin_prefetch(aRow0 + k0 + 64, 0, 1); // global_prefetch_b8
        __builtin_prefetch(bRow0 + k0 + 64, 0, 1);
        FragAB a0, a1, b0, b1;
        // ISA 16-bit A layout: elems 0..7 -> K = hl*8 + j ; elems 8..15 -> K = 16 + hl*8 + j
        a0.q[0] = *(const U4*)(aRow0 + k0 + hl * 8);
        a0.q[1] = *(const U4*)(aRow0 + k0 + 16 + hl * 8);
        a1.q[0] = *(const U4*)(aRow1 + k0 + hl * 8);
        a1.q[1] = *(const U4*)(aRow1 + k0 + 16 + hl * 8);
        b0.q[0] = *(const U4*)(bRow0 + k0 + hl * 8);
        b0.q[1] = *(const U4*)(bRow0 + k0 + 16 + hl * 8);
        b1.q[0] = *(const U4*)(bRow1 + k0 + hl * 8);
        b1.q[1] = *(const U4*)(bRow1 + k0 + 16 + hl * 8);
        acc00 = __builtin_amdgcn_wmma_f32_16x16x32_bf16(false, a0.v, false, b0.v,
                                                        (short)0, acc00, false, false);
        acc01 = __builtin_amdgcn_wmma_f32_16x16x32_bf16(false, a0.v, false, b1.v,
                                                        (short)0, acc01, false, false);
        acc10 = __builtin_amdgcn_wmma_f32_16x16x32_bf16(false, a1.v, false, b0.v,
                                                        (short)0, acc10, false, false);
        acc11 = __builtin_amdgcn_wmma_f32_16x16x32_bf16(false, a1.v, false, b1.v,
                                                        (short)0, acc11, false, false);
    }

    auto storeTile = [&](const v8f& acc, int mT, int nT) {
        const int col = nT + r;                    // C layout: VGPR i -> M = i + 8*hl, N = r
        if (col >= Nout) return;
        const float badd = bias ? bias[col] : 0.0f;
#pragma unroll
        for (int i = 0; i < 8; ++i) {
            const int row = mT + hl * 8 + i;
            if (row < M) C[(size_t)row * ldc + col] = acc[i] + badd;
        }
    };
    storeTile(acc00, mBase,      nBase);
    storeTile(acc01, mBase,      nBase + 16);
    storeTile(acc10, mBase + 16, nBase);
    storeTile(acc11, mBase + 16, nBase + 16);
}

// ---------------- GCN aggregation ----------------
// self-loop term initializes the accumulator: agg = norm[i]^2 * h[i]
__global__ void agg_init_k(const float* __restrict__ h, const float* __restrict__ nrm,
                           float* __restrict__ agg, size_t total) {
    size_t i = (size_t)blockIdx.x * blockDim.x + threadIdx.x;
    if (i >= total) return;
    float s = nrm[i >> 9];                        // node = i / 512
    agg[i] = s * s * h[i];
}
// edge scatter: agg[dst] += norm[src]*norm[dst] * h[src]; 128 threads/edge, float4 each
__global__ void agg_edges_k(const int* __restrict__ src, const int* __restrict__ dst,
                            const float* __restrict__ nrm, const float* __restrict__ h,
                            float* __restrict__ agg, int E) {
    size_t gid = (size_t)blockIdx.x * blockDim.x + threadIdx.x;
    size_t total = (size_t)E * (FEAT / 4);
    if (gid >= total) return;
    int e = (int)(gid >> 7);
    int j = (int)(gid & 127) << 2;
    int s = src[e], d = dst[e];
    float c = nrm[s] * nrm[d];
    const float4 hv = *(const float4*)(h + (size_t)s * FEAT + j);
    float* out = agg + (size_t)d * FEAT + j;
    atomicAdd(out + 0, c * hv.x);
    atomicAdd(out + 1, c * hv.y);
    atomicAdd(out + 2, c * hv.z);
    atomicAdd(out + 3, c * hv.w);
}
// bias + ReLU + convert to bf16 for the next WMMA GEMM
__global__ void bias_relu_bf16_k(const float* __restrict__ agg, const float* __restrict__ bias,
                                 unsigned short* __restrict__ outb, size_t total) {
    size_t i = (size_t)blockIdx.x * blockDim.x + threadIdx.x;
    if (i >= total) return;
    float v = agg[i] + bias[i & (FEAT - 1)];
    v = v > 0.0f ? v : 0.0f;
    outb[i] = f2bf(v);
}

// ---------------------------------------------------------------------------
extern "C" void kernel_launch(void* const* d_in, const int* in_sizes, int n_in,
                              void* d_out, int out_size, void* d_ws, size_t ws_size,
                              hipStream_t stream) {
    const float* x  = (const float*)d_in[0];
    const int*   ei = (const int*)  d_in[1];
    const float* W1 = (const float*)d_in[2];
    const float* b1 = (const float*)d_in[3];
    const float* W2 = (const float*)d_in[4];
    const float* b2 = (const float*)d_in[5];
    const float* Wc = (const float*)d_in[6];
    const float* bc = (const float*)d_in[7];

    const int nNodes    = in_sizes[0] / FEAT;     // 50000
    const int nEdges    = in_sizes[1] / 2;        // 500000
    const int nClasses  = in_sizes[7];            // 100
    const int nClassPad = 128;                    // 8 tiles of 16
    const int* src = ei;
    const int* dst = ei + nEdges;

    // ---- workspace carve (all 256B aligned) ----
    char* p = (char*)d_ws;
    auto carve = [&](size_t bytes) -> void* {
        void* r = (void*)p; p += (bytes + 255) & ~(size_t)255; return r;
    };
    float*          deg = (float*)         carve((size_t)nNodes * 4);
    float*          nrm = (float*)         carve((size_t)nNodes * 4);
    unsigned short* xb  = (unsigned short*)carve((size_t)nNodes * FEAT * 2);
    unsigned short* w1b = (unsigned short*)carve((size_t)FEAT * FEAT * 2);
    unsigned short* w2b = (unsigned short*)carve((size_t)FEAT * FEAT * 2);
    unsigned short* wcb = (unsigned short*)carve((size_t)nClassPad * FEAT * 2);
    float*          hF  = (float*)         carve((size_t)nNodes * FEAT * 4);  // GEMM out
    float*          agg = (float*)         carve((size_t)nNodes * FEAT * 4);  // aggregation
    unsigned short* hb  = (unsigned short*)carve((size_t)nNodes * FEAT * 2);  // relu'd bf16

    const size_t nodeFeat = (size_t)nNodes * FEAT;
    const int TB = 256;
    auto blocks = [&](size_t n) { return (unsigned int)((n + TB - 1) / TB); };

    // ---- normalization coefficients ----
    deg_init_k <<<blocks(nNodes), TB, 0, stream>>>(deg, nNodes);
    deg_edges_k<<<blocks(nEdges), TB, 0, stream>>>(dst, deg, nEdges);
    norm_k     <<<blocks(nNodes), TB, 0, stream>>>(deg, nrm, nNodes);

    // ---- bf16 conversions ----
    f32_to_bf16_k<<<blocks(nodeFeat), TB, 0, stream>>>(x, xb, nodeFeat);
    conv_pad_k<<<blocks((size_t)FEAT * FEAT), TB, 0, stream>>>(W1, w1b, FEAT, FEAT, FEAT);
    conv_pad_k<<<blocks((size_t)FEAT * FEAT), TB, 0, stream>>>(W2, w2b, FEAT, FEAT, FEAT);
    conv_pad_k<<<blocks((size_t)nClassPad * FEAT), TB, 0, stream>>>(Wc, wcb, nClasses, FEAT, nClassPad);

    const unsigned int mBlocks = (unsigned int)((nNodes + 63) / 64);          // 782
    const dim3 gemmGridL(mBlocks, FEAT / 128);                                // 782 x 4
    const dim3 gemmGridC(mBlocks, nClassPad / 128);                           // 782 x 1
    const size_t edgeWork = (size_t)nEdges * (FEAT / 4);

    // ---- layer 1 ----
    gemm_bf16_wmma_k<<<gemmGridL, TB, 0, stream>>>(xb, w1b, hF, nullptr,
                                                   nNodes, FEAT, FEAT, FEAT);
    agg_init_k <<<blocks(nodeFeat), TB, 0, stream>>>(hF, nrm, agg, nodeFeat);
    agg_edges_k<<<blocks(edgeWork), TB, 0, stream>>>(src, dst, nrm, hF, agg, nEdges);
    bias_relu_bf16_k<<<blocks(nodeFeat), TB, 0, stream>>>(agg, b1, hb, nodeFeat);

    // ---- layer 2 ----
    gemm_bf16_wmma_k<<<gemmGridL, TB, 0, stream>>>(hb, w2b, hF, nullptr,
                                                   nNodes, FEAT, FEAT, FEAT);
    agg_init_k <<<blocks(nodeFeat), TB, 0, stream>>>(hF, nrm, agg, nodeFeat);
    agg_edges_k<<<blocks(edgeWork), TB, 0, stream>>>(src, dst, nrm, hF, agg, nEdges);
    bias_relu_bf16_k<<<blocks(nodeFeat), TB, 0, stream>>>(agg, b2, hb, nodeFeat);

    // ---- classifier: write d_out [nNodes, nClasses] directly with bias ----
    gemm_bf16_wmma_k<<<gemmGridC, TB, 0, stream>>>(hb, wcb, (float*)d_out, bc,
                                                   nNodes, FEAT, nClasses, nClasses);
}
